// GraphLayer_17746804867118
// MI455X (gfx1250) — compile-verified, matrix operand order checked by reading, not run on hardware
//
#include <hip/hip_runtime.h>
#include <hip/hip_bf16.h>
#include <math.h>

typedef __attribute__((ext_vector_type(16))) _Float16 v16h;
typedef __attribute__((ext_vector_type(8)))  float    v8f;

static constexpr int   DD        = 128;   // feature dim
static constexpr float NEG_SLOPE = 0.2f;
static constexpr float LN_EPS    = 1e-5f;

// ---------------------------------------------------------------------------
// float atomic max via int/uint punning (monotone bit trick, works with -inf)
// ---------------------------------------------------------------------------
__device__ __forceinline__ void atomicMaxF32(float* addr, float val) {
    if (val >= 0.0f) atomicMax((int*)addr, __float_as_int(val));
    else             atomicMin((unsigned int*)addr, __float_as_uint(val));
}

// 128-thread (4-wave) block sum reduce; smem must hold >= 4 floats
__device__ __forceinline__ float blk_reduce(float v, float* smem) {
    #pragma unroll
    for (int off = 16; off > 0; off >>= 1) v += __shfl_xor(v, off, 32);
    int wave = threadIdx.x >> 5;
    if ((threadIdx.x & 31) == 0) smem[wave] = v;
    __syncthreads();
    float r = smem[0] + smem[1] + smem[2] + smem[3];
    __syncthreads();
    return r;
}

// ---------------------------------------------------------------------------
// K0: convert x -> f16, transpose+convert weights, zero/neg-inf init buffers
// ---------------------------------------------------------------------------
__global__ void k_init(const float* __restrict__ x,
                       const float* __restrict__ gat_W,
                       const float* __restrict__ gcn_W,
                       _Float16* __restrict__ xh,
                       _Float16* __restrict__ WgT,   // [n][k] col-major of gat_W
                       _Float16* __restrict__ WcT,   // [n][k] col-major of gcn_W
                       float* __restrict__ gat_acc,
                       float* __restrict__ gcn_acc,
                       float* __restrict__ e_max,
                       float* __restrict__ w_sum,
                       float* __restrict__ deg,
                       int n_nodes)
{
    long long total  = (long long)n_nodes * DD;
    long long stride = (long long)gridDim.x * blockDim.x;
    for (long long i = (long long)blockIdx.x * blockDim.x + threadIdx.x;
         i < total; i += stride) {
        xh[i]      = (_Float16)x[i];
        gat_acc[i] = 0.0f;
        gcn_acc[i] = 0.0f;
        if (i < DD * DD) {
            int nn = (int)(i >> 7);      // output column
            int kk = (int)(i & 127);     // k index
            WgT[i] = (_Float16)gat_W[kk * DD + nn];
            WcT[i] = (_Float16)gcn_W[kk * DD + nn];
        }
        if (i < (long long)n_nodes * 4) {
            e_max[i] = -__builtin_inff();
            w_sum[i] = 0.0f;
        }
        if (i < n_nodes) deg[i] = 0.0f;
    }
}

// ---------------------------------------------------------------------------
// K1: C[M,128] = A[M,128](f16,row-major) @ B(col-major f16 Bt[n*128+k])
// One wave per 16-row tile; 8 column tiles per wave; K=128 as 4 x WMMA k=32.
// Fragment layouts per CDNA5 ISA 7.12.2:
//   A 16x32 f16 : lane l (m = l&15, lg = l>>4) holds K = lg*8+{0..7} and
//                 16+lg*8+{0..7}  (pairs contiguous in K -> dword loads)
//   B 32x16 f16 : mirrored, n = l&15
//   D 16x16 f32 : vgpr v, lane l -> M = v + 8*lg, N = l&15
// ---------------------------------------------------------------------------
__global__ void __launch_bounds__(256)
k_wmma_gemm(const _Float16* __restrict__ A,
            const _Float16* __restrict__ Bt,
            float* __restrict__ C,
            int M)
{
    const int lane = threadIdx.x & 31;
    const int wave = threadIdx.x >> 5;
    const int Mt   = (M + 15) >> 4;
    const int mt   = blockIdx.x * (blockDim.x >> 5) + wave;
    if (mt >= Mt) return;                      // wave-uniform: EXEC all-ones below

    const int lg = lane >> 4;
    const int ln = lane & 15;
    int rowA = mt * 16 + ln;
    if (rowA >= M) rowA = M - 1;               // clamp keeps EXEC all-ones

    v8f acc[8];
    #pragma unroll
    for (int nt = 0; nt < 8; ++nt)
        #pragma unroll
        for (int v = 0; v < 8; ++v) acc[nt][v] = 0.0f;

    #pragma unroll
    for (int ks = 0; ks < 4; ++ks) {
        v16h a;
        const _Float16* ap = A + (size_t)rowA * DD + ks * 32 + lg * 8;
        #pragma unroll
        for (int i = 0; i < 8; ++i) { a[i] = ap[i]; a[8 + i] = ap[16 + i]; }

        #pragma unroll
        for (int nt = 0; nt < 8; ++nt) {
            v16h b;
            const _Float16* bp = Bt + (size_t)(nt * 16 + ln) * DD + ks * 32 + lg * 8;
            #pragma unroll
            for (int i = 0; i < 8; ++i) { b[i] = bp[i]; b[8 + i] = bp[16 + i]; }
            acc[nt] = __builtin_amdgcn_wmma_f32_16x16x32_f16(
                false, a, false, b, (short)0, acc[nt], false, false);
        }
    }

    #pragma unroll
    for (int nt = 0; nt < 8; ++nt)
        #pragma unroll
        for (int v = 0; v < 8; ++v) {
            int r = mt * 16 + v + 8 * lg;
            if (r < M) C[(size_t)r * DD + nt * 16 + ln] = acc[nt][v];
        }
}

// ---------------------------------------------------------------------------
// K2: a_src[n,h] = sum_c h[n,h,c]*att_src[h,c]; likewise a_dst.
// grid = n_nodes, block = 128 (wave == head), wave32 shuffle reduce.
// ---------------------------------------------------------------------------
__global__ void k_node_att(const float* __restrict__ h,
                           const float* __restrict__ att_src,
                           const float* __restrict__ att_dst,
                           float* __restrict__ a_src,
                           float* __restrict__ a_dst)
{
    const int n    = blockIdx.x;
    const int head = threadIdx.x >> 5;
    const int c    = threadIdx.x & 31;
    float hv = h[(size_t)n * DD + head * 32 + c];
    float ps = hv * att_src[head * 32 + c];
    float pd = hv * att_dst[head * 32 + c];
    #pragma unroll
    for (int off = 16; off > 0; off >>= 1) {
        ps += __shfl_xor(ps, off, 32);
        pd += __shfl_xor(pd, off, 32);
    }
    if (c == 0) {
        a_src[n * 4 + head] = ps;
        a_dst[n * 4 + head] = pd;
    }
}

// ---------------------------------------------------------------------------
// K3: per-edge LeakyReLU(e), segment-max via atomicMaxF32, degree count.
// Edges [0,nE) from edge_index; [nE, Et) are self-loops.
// ---------------------------------------------------------------------------
__global__ void k_edge_max(const int* __restrict__ src, const int* __restrict__ dst,
                           long long nE, long long Et,
                           const float* __restrict__ a_src, const float* __restrict__ a_dst,
                           float* __restrict__ e_buf, float* __restrict__ e_max,
                           float* __restrict__ deg)
{
    long long e = (long long)blockIdx.x * blockDim.x + threadIdx.x;
    if (e >= Et) return;
    int s, t;
    if (e < nE) { s = src[e]; t = dst[e]; } else { s = t = (int)(e - nE); }
    float4 as = ((const float4*)a_src)[s];
    float4 ad = ((const float4*)a_dst)[t];
    float ev[4] = { as.x + ad.x, as.y + ad.y, as.z + ad.z, as.w + ad.w };
    #pragma unroll
    for (int hh = 0; hh < 4; ++hh) {
        float v = ev[hh];
        v = v > 0.0f ? v : NEG_SLOPE * v;
        e_buf[e * 4 + hh] = v;
        atomicMaxF32(&e_max[(size_t)t * 4 + hh], v);
    }
    atomicAdd(&deg[t], 1.0f);
    (void)s;
}

// ---------------------------------------------------------------------------
// K4: w = exp(e - max); store in place; accumulate softmax denominators.
// ---------------------------------------------------------------------------
__global__ void k_edge_w(const int* __restrict__ src, const int* __restrict__ dst,
                         long long nE, long long Et,
                         float* __restrict__ e_buf, const float* __restrict__ e_max,
                         float* __restrict__ w_sum)
{
    long long e = (long long)blockIdx.x * blockDim.x + threadIdx.x;
    if (e >= Et) return;
    int t;
    if (e < nE) { t = dst[e]; } else { t = (int)(e - nE); }
    #pragma unroll
    for (int hh = 0; hh < 4; ++hh) {
        float w = __expf(e_buf[e * 4 + hh] - e_max[(size_t)t * 4 + hh]);
        e_buf[e * 4 + hh] = w;
        atomicAdd(&w_sum[(size_t)t * 4 + hh], w);
    }
    (void)src;
}

// K5: dinv = deg^-0.5 (self-loops guarantee deg >= 1)
__global__ void k_dinv(const float* __restrict__ deg, float* __restrict__ dinv, int n_nodes)
{
    int n = blockIdx.x * blockDim.x + threadIdx.x;
    if (n >= n_nodes) return;
    float d = deg[n];
    dinv[n] = d > 0.0f ? rsqrtf(d) : 0.0f;
}

// ---------------------------------------------------------------------------
// K6: fused GAT + GCN message scatter. 128 threads per edge (one per channel).
// ---------------------------------------------------------------------------
__global__ void k_edge_scatter(const int* __restrict__ src, const int* __restrict__ dst,
                               long long nE,
                               const float* __restrict__ e_buf, const float* __restrict__ w_sum,
                               const float* __restrict__ dinv,
                               const float* __restrict__ h, const float* __restrict__ g,
                               float* __restrict__ gat_acc, float* __restrict__ gcn_acc)
{
    long long e = blockIdx.x;
    int s, t;
    if (e < nE) { s = src[e]; t = dst[e]; } else { s = t = (int)(e - nE); }
    const int d    = threadIdx.x;
    const int head = d >> 5;
    float alpha = e_buf[e * 4 + head] / w_sum[(size_t)t * 4 + head];
    float norm  = dinv[s] * dinv[t];
    atomicAdd(&gat_acc[(size_t)t * DD + d], alpha * h[(size_t)s * DD + d]);
    atomicAdd(&gcn_acc[(size_t)t * DD + d], norm  * g[(size_t)s * DD + d]);
}

// ---------------------------------------------------------------------------
// K7: bias + gate softmax + residual + LayerNorm. grid = n_nodes, block = 128.
// ---------------------------------------------------------------------------
__global__ void k_fuse_ln(const float* __restrict__ x,
                          const float* __restrict__ gat_acc, const float* __restrict__ gcn_acc,
                          const float* __restrict__ gat_bias, const float* __restrict__ gcn_bias,
                          const float* __restrict__ gate_W, const float* __restrict__ gate_b,
                          const float* __restrict__ gamma, const float* __restrict__ beta,
                          float* __restrict__ out)
{
    __shared__ float smem[4];
    const int n = blockIdx.x;
    const int d = threadIdx.x;
    float a = gat_acc[(size_t)n * DD + d] + gat_bias[d];
    float b = gcn_acc[(size_t)n * DD + d] + gcn_bias[d];

    float z0 = a * gate_W[d * 2 + 0] + b * gate_W[(DD + d) * 2 + 0];
    float z1 = a * gate_W[d * 2 + 1] + b * gate_W[(DD + d) * 2 + 1];
    z0 = blk_reduce(z0, smem) + gate_b[0];
    z1 = blk_reduce(z1, smem) + gate_b[1];

    float m  = fmaxf(z0, z1);
    float e0 = __expf(z0 - m), e1 = __expf(z1 - m);
    float inv = 1.0f / (e0 + e1);
    float y  = (e0 * inv) * a + (e1 * inv) * b + x[(size_t)n * DD + d];

    float mu  = blk_reduce(y, smem) * (1.0f / DD);
    float dv  = y - mu;
    float var = blk_reduce(dv * dv, smem) * (1.0f / DD);
    out[(size_t)n * DD + d] = dv * rsqrtf(var + LN_EPS) * gamma[d] + beta[d];
}

// ---------------------------------------------------------------------------
extern "C" void kernel_launch(void* const* d_in, const int* in_sizes, int n_in,
                              void* d_out, int out_size, void* d_ws, size_t ws_size,
                              hipStream_t stream) {
    const float* x        = (const float*)d_in[0];
    const float* gat_W    = (const float*)d_in[1];
    const float* att_src  = (const float*)d_in[2];
    const float* att_dst  = (const float*)d_in[3];
    const float* gat_bias = (const float*)d_in[4];
    const float* gcn_W    = (const float*)d_in[5];
    const float* gcn_bias = (const float*)d_in[6];
    const float* gate_W   = (const float*)d_in[7];
    const float* gate_b   = (const float*)d_in[8];
    const float* ln_gamma = (const float*)d_in[9];
    const float* ln_beta  = (const float*)d_in[10];
    const int*   ei       = (const int*)d_in[11];   // int32 (JAX default x64-off)

    const int       n_nodes = in_sizes[0] / DD;
    const long long nE      = in_sizes[11] / 2;
    const long long Et      = nE + n_nodes;
    const int* src = ei;
    const int* dst = ei + nE;

    // workspace carve-up (256B aligned)
    char*  ws  = (char*)d_ws;
    size_t off = 0;
    auto alloc = [&](size_t bytes) -> void* {
        void* p = ws + off;
        off += (bytes + 255) & ~(size_t)255;
        return p;
    };
    _Float16* xh      = (_Float16*)alloc((size_t)n_nodes * DD * 2);
    _Float16* WgT     = (_Float16*)alloc((size_t)DD * DD * 2);
    _Float16* WcT     = (_Float16*)alloc((size_t)DD * DD * 2);
    float*    h       = (float*)alloc((size_t)n_nodes * DD * 4);
    float*    g       = (float*)alloc((size_t)n_nodes * DD * 4);
    float*    gat_acc = (float*)alloc((size_t)n_nodes * DD * 4);
    float*    gcn_acc = (float*)alloc((size_t)n_nodes * DD * 4);
    float*    a_src   = (float*)alloc((size_t)n_nodes * 4 * 4);
    float*    a_dst   = (float*)alloc((size_t)n_nodes * 4 * 4);
    float*    e_max   = (float*)alloc((size_t)n_nodes * 4 * 4);
    float*    w_sum   = (float*)alloc((size_t)n_nodes * 4 * 4);
    float*    deg     = (float*)alloc((size_t)n_nodes * 4);
    float*    dinv    = (float*)alloc((size_t)n_nodes * 4);
    float*    e_buf   = (float*)alloc((size_t)Et * 4 * 4);

    // K0: init / convert
    {
        long long total = (long long)n_nodes * DD;
        int blocks = (int)((total + 255) / 256);
        k_init<<<blocks, 256, 0, stream>>>(x, gat_W, gcn_W, xh, WgT, WcT,
                                           gat_acc, gcn_acc, e_max, w_sum, deg, n_nodes);
    }
    // K1: WMMA GEMMs  h = x@gat_W, g = x@gcn_W
    {
        int Mt = (n_nodes + 15) / 16;
        int blocks = (Mt + 7) / 8;                 // 8 waves per 256-thread block
        k_wmma_gemm<<<blocks, 256, 0, stream>>>(xh, WgT, h, n_nodes);
        k_wmma_gemm<<<blocks, 256, 0, stream>>>(xh, WcT, g, n_nodes);
    }
    // K2: attention coefficients
    k_node_att<<<n_nodes, 128, 0, stream>>>(h, att_src, att_dst, a_src, a_dst);
    // K3: edge e + segment max + degree
    {
        int blocks = (int)((Et + 255) / 256);
        k_edge_max<<<blocks, 256, 0, stream>>>(src, dst, nE, Et, a_src, a_dst,
                                               e_buf, e_max, deg);
        // K4: exp weights + denominators
        k_edge_w<<<blocks, 256, 0, stream>>>(src, dst, nE, Et, e_buf, e_max, w_sum);
    }
    // K5: deg^-0.5
    k_dinv<<<(n_nodes + 255) / 256, 256, 0, stream>>>(deg, dinv, n_nodes);
    // K6: fused GAT+GCN scatter (one block per edge incl. self-loops)
    k_edge_scatter<<<(unsigned)Et, 128, 0, stream>>>(src, dst, nE, e_buf, w_sum, dinv,
                                                     h, g, gat_acc, gcn_acc);
    // K7: gate + residual + LayerNorm
    k_fuse_ln<<<n_nodes, 128, 0, stream>>>(x, gat_acc, gcn_acc, gat_bias, gcn_bias,
                                           gate_W, gate_b, ln_gamma, ln_beta,
                                           (float*)d_out);
    (void)n_in; (void)out_size; (void)ws_size;
}
